// POINTNET2_71932112273929
// MI455X (gfx1250) — matrix operand order repaired and checked by reading, not compiled, not run on hardware
//
#include <hip/hip_runtime.h>
#include <math.h>

// ---------------------------------------------------------------------------
// PointNet++ segmentation forward for MI455X (gfx1250, wave32).
// All pointwise MLPs are f16 WMMA GEMMs (v_wmma_f32_16x16x32_f16) with
// bias+BN folded to per-channel scale/shift and ReLU in the epilogue.
// Activations are f16 end-to-end; ragged K is zero-padded to 32 at tensor
// construction so the GEMM has no bounds checks. The K-loop is explicitly
// double-buffered: loads for step k+32 are issued before the WMMAs of step k,
// so b128 loads pipeline under matrix issue instead of draining loadcnt.
// ---------------------------------------------------------------------------

typedef __attribute__((ext_vector_type(8)))  _Float16 v8h;
typedef __attribute__((ext_vector_type(16))) _Float16 v16h;
typedef __attribute__((ext_vector_type(8)))  float    v8f;
typedef _Float16 h16;

#define NSAMPLE 32

// ------------------------- Farthest point sampling -------------------------
__global__ void fps_kernel(const float* __restrict__ xyz, int N, int npoint,
                           int* __restrict__ fps_idx, float* __restrict__ dist_ws) {
  const int b   = blockIdx.x;
  const int tid = threadIdx.x;                 // 256 threads
  const float* p = xyz + (size_t)b * N * 3;
  float* dist = dist_ws + (size_t)b * N;

  __shared__ float s_val[256];
  __shared__ int   s_idx[256];
  __shared__ float cx, cy, cz;
  __shared__ int   s_far;

  for (int i = tid; i < N; i += 256) dist[i] = 1e10f;
  if (tid == 0) s_far = 0;
  __syncthreads();

  for (int it = 0; it < npoint; ++it) {
    if (tid == 0) {
      int far = s_far;
      fps_idx[b * npoint + it] = far;
      cx = p[far * 3 + 0]; cy = p[far * 3 + 1]; cz = p[far * 3 + 2];
    }
    __syncthreads();
    float best = -1.0f; int bi = 0;
    for (int i = tid; i < N; i += 256) {
      float dx = p[i * 3 + 0] - cx;
      float dy = p[i * 3 + 1] - cy;
      float dz = p[i * 3 + 2] - cz;
      float d  = dx * dx + dy * dy + dz * dz;
      float nd = fminf(dist[i], d);
      dist[i] = nd;
      if (nd > best) { best = nd; bi = i; }    // strict > keeps first max
    }
    s_val[tid] = best; s_idx[tid] = bi;
    __syncthreads();
    for (int s = 128; s > 0; s >>= 1) {
      if (tid < s) {
        bool take = (s_val[tid + s] > s_val[tid]) ||
                    (s_val[tid + s] == s_val[tid] && s_idx[tid + s] < s_idx[tid]);
        if (take) { s_val[tid] = s_val[tid + s]; s_idx[tid] = s_idx[tid + s]; }
      }
      __syncthreads();
    }
    if (tid == 0) s_far = s_idx[0];
    __syncthreads();
  }
}

// ------------------------------- Gathers -----------------------------------
__global__ void gather_xyz_kernel(const float* __restrict__ xyz, const int* __restrict__ fidx,
                                  int N, int S, float* __restrict__ out, int B) {
  int t = blockIdx.x * blockDim.x + threadIdx.x;
  if (t >= B * S) return;
  int b = t / S;
  int j = fidx[t];
  const float* p = xyz + ((size_t)b * N + j) * 3;
  float* q = out + (size_t)t * 3;
  q[0] = p[0]; q[1] = p[1]; q[2] = p[2];
}

// f32 -> f16 convert (for l0 features = xyz)
__global__ void cvt_f16_kernel(const float* __restrict__ in, h16* __restrict__ out, int n) {
  int t = blockIdx.x * blockDim.x + threadIdx.x;
  if (t < n) out[t] = (h16)in[t];
}

// ------------------------------ Ball query ---------------------------------
__global__ void ball_query_kernel(const float* __restrict__ xyz, const float* __restrict__ qxyz,
                                  int N, int S, float r2, int* __restrict__ idx, int B) {
  int t = blockIdx.x * blockDim.x + threadIdx.x;
  if (t >= B * S) return;
  int b = t / S;
  const float* p = xyz + (size_t)b * N * 3;
  float qx = qxyz[(size_t)t * 3 + 0];
  float qy = qxyz[(size_t)t * 3 + 1];
  float qz = qxyz[(size_t)t * 3 + 2];
  int* out = idx + (size_t)t * NSAMPLE;
  int cnt = 0, first = N - 1;
  for (int j = 0; j < N && cnt < NSAMPLE; ++j) {
    float dx = p[j * 3 + 0] - qx;
    float dy = p[j * 3 + 1] - qy;
    float dz = p[j * 3 + 2] - qz;
    float d  = dx * dx + dy * dy + dz * dz;
    if (d <= r2) { if (cnt == 0) first = j; out[cnt++] = j; }
  }
  for (; cnt < NSAMPLE; ++cnt) out[cnt] = first;
}

// -------------------------- Grouping (concat, f16, padded) -----------------
// grouped[row, 0:3] = xyz[idx]-new_xyz ; [3:3+Cpts] = points[idx] ; tail = 0
__global__ void group_kernel(const float* __restrict__ xyz, const float* __restrict__ qxyz,
                             const h16* __restrict__ points, int N, int S, int Cpts, int Kp,
                             h16* __restrict__ grouped, const int* __restrict__ idx, int B) {
  int t = blockIdx.x * blockDim.x + threadIdx.x;
  if (t >= B * S * NSAMPLE) return;
  int s = (t / NSAMPLE) % S;
  int b =  t / (NSAMPLE * S);
  int j = idx[t];
  h16* g = grouped + (size_t)t * Kp;
  const float* pj = xyz + ((size_t)b * N + j) * 3;
  const float* q  = qxyz + ((size_t)b * S + s) * 3;
  g[0] = (h16)(pj[0] - q[0]); g[1] = (h16)(pj[1] - q[1]); g[2] = (h16)(pj[2] - q[2]);
  const h16* f = points + ((size_t)b * N + j) * Cpts;
  for (int c = 0; c < Cpts; ++c) g[3 + c] = f[c];
  for (int c = 3 + Cpts; c < Kp; ++c) g[c] = (h16)0.0f;
}

// ----------------- Per-layer prep: W f32 -> f16 padded, BN fold ------------
// sc = gamma*rsqrt(var+eps) ; sh = beta + (bias-mean)*sc ; y = relu(acc*sc+sh)
__global__ void prep_kernel(const float* __restrict__ W, const float* __restrict__ b,
                            const float* __restrict__ g, const float* __restrict__ be,
                            const float* __restrict__ mu, const float* __restrict__ var,
                            h16* __restrict__ Wh, float* __restrict__ sc, float* __restrict__ sh,
                            int CO, int K, int Kp) {
  int t = blockIdx.x * blockDim.x + threadIdx.x;
  if (t < CO * Kp) {
    int co = t / Kp, k = t % Kp;
    Wh[t] = (h16)((k < K) ? W[(size_t)co * K + k] : 0.0f);
  }
  if (t < CO) {
    float s = g[t] * rsqrtf(var[t] + 1e-5f);
    sc[t] = s;
    sh[t] = be[t] + (b[t] - mu[t]) * s;
  }
}

// ------------------- Fused GEMM + BN + ReLU (WMMA, f16) --------------------
// Y[M,CO] = relu((X[M,Kp] * Wh[CO,Kp]^T) * sc + sh), M%16==0, Kp%32==0.
// One wave computes a 16 x (16*NT) strip. Double-buffered K-loop: fragments
// for step k+32 are loaded before the WMMAs of step k execute.
template <int NT>
__global__ void mlp_wmma_kernel(const h16* __restrict__ X, const h16* __restrict__ Wh,
                                const float* __restrict__ sc, const float* __restrict__ sh,
                                h16* __restrict__ Y, int M, int Kp, int CO) {
  const int lane = threadIdx.x & 31;
  const int wave = threadIdx.x >> 5;                 // 0..7
  const int r    = lane & 15;
  const int hi   = lane >> 4;
  const int ntn  = CO / (16 * NT);
  const long long mtiles = M >> 4;
  const long long tile   = (long long)blockIdx.x * 8 + wave;
  if (tile >= mtiles * ntn) return;
  const int tm   = (int)(tile / ntn);
  const int tn   = (int)(tile % ntn);
  const int col0 = tn * 16 * NT;

  v8f acc[NT];
#pragma unroll
  for (int t = 0; t < NT; ++t) acc[t] = (v8f){};

  // A k order used here (contiguous, same as B): lane half 0 -> K {0..15}, half 1 -> {16..31}
  const h16* xrow = X + (size_t)(tm * 16 + r) * Kp;
  // B layout: lane col n = r, K = hi*16 + 0..15 packed sequentially
  const h16* wr0 = Wh + (size_t)(col0 + r) * Kp + hi * 16;

  auto loadA = [&](int k0) -> v16h {
    v8h a0 = *(const v8h*)(xrow + k0 + hi * 16);
    v8h a1 = *(const v8h*)(xrow + k0 + hi * 16 + 8);
    return __builtin_shufflevector(a0, a1, 0, 1, 2, 3, 4, 5, 6, 7,
                                           8, 9, 10, 11, 12, 13, 14, 15);
  };
  auto loadB = [&](int t, int k0) -> v16h {
    const h16* wrow = wr0 + (size_t)t * 16 * Kp + k0;
    v8h b0 = *(const v8h*)(wrow);
    v8h b1 = *(const v8h*)(wrow + 8);
    return __builtin_shufflevector(b0, b1, 0, 1, 2, 3, 4, 5, 6, 7,
                                           8, 9, 10, 11, 12, 13, 14, 15);
  };

  v16h a = loadA(0);
  v16h bv[NT];
#pragma unroll
  for (int t = 0; t < NT; ++t) bv[t] = loadB(t, 0);

  int k0 = 0;
  for (; k0 + 32 < Kp; k0 += 32) {
    // Prefetch next K-step into the second buffer before consuming this one.
    v16h an = loadA(k0 + 32);
    v16h bn[NT];
#pragma unroll
    for (int t = 0; t < NT; ++t) bn[t] = loadB(t, k0 + 32);
#pragma unroll
    for (int t = 0; t < NT; ++t)
      acc[t] = __builtin_amdgcn_wmma_f32_16x16x32_f16(
          false, a, false, bv[t], (short)0, acc[t], false, false);
    a = an;
#pragma unroll
    for (int t = 0; t < NT; ++t) bv[t] = bn[t];
  }
  // Peeled last K-step.
#pragma unroll
  for (int t = 0; t < NT; ++t)
    acc[t] = __builtin_amdgcn_wmma_f32_16x16x32_f16(
        false, a, false, bv[t], (short)0, acc[t], false, false);

  // Epilogue: lane owns column r of each tile; rows (hi?8..15:0..7).
#pragma unroll
  for (int t = 0; t < NT; ++t) {
    const int co = col0 + t * 16 + r;
    const float s = sc[co], h = sh[co];
#pragma unroll
    for (int i = 0; i < 8; ++i) {
      int rr = tm * 16 + (hi ? 8 + i : i);
      Y[(size_t)rr * CO + co] = (h16)fmaxf(acc[t][i] * s + h, 0.0f);
    }
  }
}

// ------------------------------ Max pooling --------------------------------
__global__ void maxpool_kernel(const h16* __restrict__ Xin, h16* __restrict__ Yout,
                               int groups, int C) {
  int t = blockIdx.x * blockDim.x + threadIdx.x;
  if (t >= groups * C) return;
  int c = t % C, g = t / C;
  const h16* p = Xin + (size_t)g * NSAMPLE * C + c;
  float m = (float)p[0];
  for (int k = 1; k < NSAMPLE; ++k) m = fmaxf(m, (float)p[(size_t)k * C]);
  Yout[(size_t)g * C + c] = (h16)m;
}

// ------------------------------ 3-NN + weights -----------------------------
__global__ void knn3_kernel(const float* __restrict__ xyz1, const float* __restrict__ xyz2,
                            int n1, int n2, int* __restrict__ kidx, float* __restrict__ kw, int B) {
  int t = blockIdx.x * blockDim.x + threadIdx.x;
  if (t >= B * n1) return;
  int b = t / n1;
  float qx = xyz1[(size_t)t * 3 + 0];
  float qy = xyz1[(size_t)t * 3 + 1];
  float qz = xyz1[(size_t)t * 3 + 2];
  const float* p = xyz2 + (size_t)b * n2 * 3;
  float d0 = 1e30f, d1 = 1e30f, d2 = 1e30f;
  int   i0 = 0,     i1 = 0,     i2 = 0;
  for (int j = 0; j < n2; ++j) {
    float dx = p[j * 3 + 0] - qx;
    float dy = p[j * 3 + 1] - qy;
    float dz = p[j * 3 + 2] - qz;
    float d  = dx * dx + dy * dy + dz * dz;
    if      (d < d0) { d2 = d1; i2 = i1; d1 = d0; i1 = i0; d0 = d; i0 = j; }
    else if (d < d1) { d2 = d1; i2 = i1; d1 = d;  i1 = j; }
    else if (d < d2) { d2 = d;  i2 = j; }
  }
  float w0 = 1.0f / (d0 + 1e-8f);
  float w1 = 1.0f / (d1 + 1e-8f);
  float w2 = 1.0f / (d2 + 1e-8f);
  float s = w0 + w1 + w2;
  kidx[(size_t)t * 3 + 0] = i0; kidx[(size_t)t * 3 + 1] = i1; kidx[(size_t)t * 3 + 2] = i2;
  kw[(size_t)t * 3 + 0] = w0 / s; kw[(size_t)t * 3 + 1] = w1 / s; kw[(size_t)t * 3 + 2] = w2 / s;
}

// ------------------- Interpolate + (optional) skip concat ------------------
__global__ void interp_concat_kernel(const h16* __restrict__ pts1, int C1,
                                     const h16* __restrict__ pts2, int C2, int Kp,
                                     const int* __restrict__ kidx, const float* __restrict__ kw,
                                     int n1, int n2, h16* __restrict__ out, int B) {
  int t = blockIdx.x * blockDim.x + threadIdx.x;
  if (t >= B * n1) return;
  int b = t / n1;
  h16* o = out + (size_t)t * Kp;
  if (pts1) for (int c = 0; c < C1; ++c) o[c] = pts1[(size_t)t * C1 + c];
  int i0 = kidx[(size_t)t * 3 + 0];
  int i1 = kidx[(size_t)t * 3 + 1];
  int i2 = kidx[(size_t)t * 3 + 2];
  float w0 = kw[(size_t)t * 3 + 0];
  float w1 = kw[(size_t)t * 3 + 1];
  float w2 = kw[(size_t)t * 3 + 2];
  const h16* p2 = pts2 + (size_t)b * n2 * C2;
  for (int c = 0; c < C2; ++c)
    o[C1 + c] = (h16)(w0 * (float)p2[(size_t)i0 * C2 + c] +
                      w1 * (float)p2[(size_t)i1 * C2 + c] +
                      w2 * (float)p2[(size_t)i2 * C2 + c]);
  for (int c = C1 + C2; c < Kp; ++c) o[c] = (h16)0.0f;
}

// ---------------- Head2 (128 -> 5) + log_softmax + transpose ---------------
__global__ void head2_kernel(const h16* __restrict__ X, const float* __restrict__ W,
                             const float* __restrict__ bias, float* __restrict__ out,
                             int N, int B, int K, int NCLS) {
  int t = blockIdx.x * blockDim.x + threadIdx.x;
  if (t >= B * N) return;
  int b = t / N, n = t % N;
  const h16* x = X + (size_t)t * K;
  float logits[8];
  for (int c = 0; c < NCLS; ++c) {
    float a = bias[c];
    const float* w = W + (size_t)c * K;
    for (int k = 0; k < K; ++k) a += (float)x[k] * w[k];
    logits[c] = a;
  }
  float mx = logits[0];
  for (int c = 1; c < NCLS; ++c) mx = fmaxf(mx, logits[c]);
  float sum = 0.0f;
  for (int c = 0; c < NCLS; ++c) sum += expf(logits[c] - mx);
  float lse = mx + logf(sum);
  for (int c = 0; c < NCLS; ++c)
    out[((size_t)b * NCLS + c) * N + n] = logits[c] - lse;   // (B, NCLS, N)
}

// ---------------------------------------------------------------------------
struct ConvBN { const float *w, *b, *g, *be, *mu, *var; };

extern "C" void kernel_launch(void* const* d_in, const int* in_sizes, int n_in,
                              void* d_out, int out_size, void* d_ws, size_t ws_size,
                              hipStream_t stream) {
  (void)in_sizes; (void)n_in; (void)out_size; (void)ws_size;
  const int B = 4, N0 = 16384, NCLS = 5;
  const float* xyz = (const float*)d_in[0];

  // conv_bn layer i occupies d_in[1+6i .. 1+6i+5] as w,b,gamma,beta,mean,var.
  // Order: sa1[0..2] sa2[3..5] sa3[6..8] sa4[9..11] fp4[12..13] fp3[14..15]
  //        fp2[16..17] fp1[18..20] head1[21]; head2 w/b = d_in[133]/d_in[134].
  auto CB = [&](int li) -> ConvBN {
    const int base = 1 + li * 6;
    return ConvBN{ (const float*)d_in[base + 0], (const float*)d_in[base + 1],
                   (const float*)d_in[base + 2], (const float*)d_in[base + 3],
                   (const float*)d_in[base + 4], (const float*)d_in[base + 5] };
  };

  // ---- workspace bump allocator (256B aligned) ----
  char* base = (char*)d_ws;
  size_t o = 0;
  auto alloc = [&](size_t bytes) -> void* {
    void* p = base + o;
    o += (bytes + 255) & ~(size_t)255;
    return p;
  };
  float* fdist = (float*)alloc((size_t)B * N0 * 4);
  float* l1x = (float*)alloc((size_t)B * 1024 * 3 * 4);
  float* l2x = (float*)alloc((size_t)B * 256 * 3 * 4);
  float* l3x = (float*)alloc((size_t)B * 64 * 3 * 4);
  float* l4x = (float*)alloc((size_t)B * 16 * 3 * 4);
  float* knnw = (float*)alloc((size_t)B * N0 * 3 * 4);
  float* scb = (float*)alloc(512 * 4);
  float* shb = (float*)alloc(512 * 4);
  int* fpsIdx  = (int*)alloc((size_t)B * 1024 * 4);
  int* ballIdx = (int*)alloc((size_t)B * 1024 * NSAMPLE * 4);
  int* knnIdx  = (int*)alloc((size_t)B * N0 * 3 * 4);
  h16* l0p = (h16*)alloc((size_t)B * N0 * 3 * 2);       // xyz as f16 features
  h16* l1p = (h16*)alloc((size_t)B * 1024 * 64 * 2);
  h16* l2p = (h16*)alloc((size_t)B * 256 * 128 * 2);
  h16* l3p = (h16*)alloc((size_t)B * 64 * 256 * 2);
  h16* l4p = (h16*)alloc((size_t)B * 16 * 512 * 2);
  h16* l3f = (h16*)alloc((size_t)B * 64 * 256 * 2);
  h16* l2f = (h16*)alloc((size_t)B * 256 * 256 * 2);
  h16* l1f = (h16*)alloc((size_t)B * 1024 * 128 * 2);
  h16* l0f = (h16*)alloc((size_t)B * N0 * 128 * 2);
  h16* Wh  = (h16*)alloc((size_t)256 * 768 * 2);        // packed layer weights
  h16* aG  = (h16*)alloc((size_t)16 * 1024 * 1024);     // grouped/concat arena
  h16* aA  = (h16*)alloc((size_t)16 * 1024 * 1024);     // MLP ping
  h16* aB  = (h16*)alloc((size_t)16 * 1024 * 1024);     // MLP pong

  auto cdiv = [](long long a, long long b) { return (int)((a + b - 1) / b); };

  auto mlp = [&](const h16* X, const ConvBN& L, h16* Y, int M, int K, int Kp, int CO) {
    prep_kernel<<<cdiv((long long)CO * Kp, 256), 256, 0, stream>>>(
        L.w, L.b, L.g, L.be, L.mu, L.var, Wh, scb, shb, CO, K, Kp);
    if (CO % 64 == 0) {
      long long tiles = (long long)(M / 16) * (CO / 64);
      mlp_wmma_kernel<4><<<cdiv(tiles, 8), 256, 0, stream>>>(X, Wh, scb, shb, Y, M, Kp, CO);
    } else {
      long long tiles = (long long)(M / 16) * (CO / 32);
      mlp_wmma_kernel<2><<<cdiv(tiles, 8), 256, 0, stream>>>(X, Wh, scb, shb, Y, M, Kp, CO);
    }
  };

  auto run_mlp = [&](const h16* X, int M, int K, int Kp0, const int* lis, const int* cos,
                     int nl, h16* dst) -> const h16* {
    const h16* cur = X; int k = K, kp = Kp0;
    for (int i = 0; i < nl; ++i) {
      h16* out2 = (i == nl - 1 && dst) ? dst : ((cur == aA) ? aB : aA);
      ConvBN L = CB(lis[i]);
      mlp(cur, L, out2, M, k, kp, cos[i]);
      cur = out2; k = cos[i]; kp = k;
    }
    return cur;
  };

  auto sa = [&](const float* ix, const h16* ip, int N, int Cpts, int S, float radius,
                const int* lis, const int* cos, float* ox, h16* op) {
    int Kp = (3 + Cpts + 31) & ~31;
    fps_kernel<<<B, 256, 0, stream>>>(ix, N, S, fpsIdx, fdist);
    gather_xyz_kernel<<<cdiv((long long)B * S, 256), 256, 0, stream>>>(ix, fpsIdx, N, S, ox, B);
    ball_query_kernel<<<cdiv((long long)B * S, 256), 256, 0, stream>>>(
        ix, ox, N, S, radius * radius, ballIdx, B);
    group_kernel<<<cdiv((long long)B * S * NSAMPLE, 256), 256, 0, stream>>>(
        ix, ox, ip, N, S, Cpts, Kp, aG, ballIdx, B);
    const h16* f = run_mlp(aG, B * S * NSAMPLE, 3 + Cpts, Kp, lis, cos, 3, nullptr);
    maxpool_kernel<<<cdiv((long long)B * S * cos[2], 256), 256, 0, stream>>>(f, op, B * S, cos[2]);
  };

  auto fp = [&](const float* x1, const float* x2, const h16* p1, int C1,
                const h16* p2, int C2, int n1, int n2,
                const int* lis, const int* cos, int nl, h16* dst) {
    int Kp = (C1 + C2 + 31) & ~31;   // already a multiple of 32 in this net
    knn3_kernel<<<cdiv((long long)B * n1, 256), 256, 0, stream>>>(x1, x2, n1, n2, knnIdx, knnw, B);
    interp_concat_kernel<<<cdiv((long long)B * n1, 256), 256, 0, stream>>>(
        p1, C1, p2, C2, Kp, knnIdx, knnw, n1, n2, aG, B);
    run_mlp(aG, B * n1, C1 + C2, Kp, lis, cos, nl, dst);
  };

  // l0 features = xyz itself, as f16
  cvt_f16_kernel<<<cdiv((long long)B * N0 * 3, 256), 256, 0, stream>>>(xyz, l0p, B * N0 * 3);

  // ---- set abstraction ----
  { int lis[3] = {0, 1, 2};    int cos[3] = {32, 32, 64};    sa(xyz, l0p, N0,   3,   1024, 0.1f, lis, cos, l1x, l1p); }
  { int lis[3] = {3, 4, 5};    int cos[3] = {64, 64, 128};   sa(l1x, l1p, 1024, 64,  256,  0.2f, lis, cos, l2x, l2p); }
  { int lis[3] = {6, 7, 8};    int cos[3] = {128, 128, 256}; sa(l2x, l2p, 256,  128, 64,   0.4f, lis, cos, l3x, l3p); }
  { int lis[3] = {9, 10, 11};  int cos[3] = {256, 256, 512}; sa(l3x, l3p, 64,   256, 16,   0.8f, lis, cos, l4x, l4p); }

  // ---- feature propagation ----
  { int lis[2] = {12, 13}; int cos[2] = {256, 256}; fp(l3x, l4x, l3p, 256, l4p, 512, 64,   16,   lis, cos, 2, l3f); }
  { int lis[2] = {14, 15}; int cos[2] = {256, 256}; fp(l2x, l3x, l2p, 128, l3f, 256, 256,  64,   lis, cos, 2, l2f); }
  { int lis[2] = {16, 17}; int cos[2] = {256, 128}; fp(l1x, l2x, l1p, 64,  l2f, 256, 1024, 256,  lis, cos, 2, l1f); }
  { int lis[3] = {18, 19, 20}; int cos[3] = {128, 128, 128}; fp(xyz, l1x, nullptr, 0, l1f, 128, N0, 1024, lis, cos, 3, l0f); }

  // ---- head ----
  { ConvBN H = CB(21); mlp(l0f, H, aA, B * N0, 128, 128, 128); }
  head2_kernel<<<cdiv((long long)B * N0, 256), 256, 0, stream>>>(
      aA, (const float*)d_in[133], (const float*)d_in[134], (float*)d_out, N0, B, 128, NCLS);
}
